// GATNet_1486058684431
// MI455X (gfx1250) — matrix operand order repaired and checked by reading, not compiled, not run on hardware
//
#include <hip/hip_runtime.h>
#include <hip/hip_bf16.h>

// ---------------------------------------------------------------------------
// GATv2 6-layer forward for gfx1250 (MI455X).
// Dense GEMMs: V_WMMA_F32_16X16X32_F16. One wave -> 64x32 f32 output
// (4 M-tiles x 2 N-tiles, 8 WMMAs per 32-wide K chunk). fp32->f16 conversion
// in registers, fp32 accumulate. Tile guards are scalarized via readfirstlane
// so EXEC stays all-1s (no saveexec juggling) at every WMMA. K-guards exist
// only in the remainder chunk (layer 0, K=2); the N edge uses clamped loads +
// masked stores. Edge stage: b128 gathers + f32 atomics (memory/atomic bound).
// ---------------------------------------------------------------------------

typedef __attribute__((ext_vector_type(16))) _Float16 v16h;
typedef __attribute__((ext_vector_type(8)))  float    v8f;

#define GAT_N  50000
#define GAT_E  600000
#define GAT_E2 650000   // E + N self loops

__device__ __forceinline__ v16h cvt_a_fast(const float* __restrict__ ap)
{
    // A f16 16x32 layout (ISA 7.12.2): this lane supplies two contiguous runs
    // of 8 K-values: [k0+hi*8 .. +7] and [k0+16+hi*8 .. +7]. 4x b128 loads.
    const float4 a0 = *(const float4*)(ap);
    const float4 a1 = *(const float4*)(ap + 4);
    const float4 a2 = *(const float4*)(ap + 16);
    const float4 a3 = *(const float4*)(ap + 20);
    v16h a;
    a[0]=(_Float16)a0.x; a[1]=(_Float16)a0.y; a[2]=(_Float16)a0.z; a[3]=(_Float16)a0.w;
    a[4]=(_Float16)a1.x; a[5]=(_Float16)a1.y; a[6]=(_Float16)a1.z; a[7]=(_Float16)a1.w;
    a[8]=(_Float16)a2.x; a[9]=(_Float16)a2.y; a[10]=(_Float16)a2.z; a[11]=(_Float16)a2.w;
    a[12]=(_Float16)a3.x; a[13]=(_Float16)a3.y; a[14]=(_Float16)a3.z; a[15]=(_Float16)a3.w;
    return a;
}

__device__ __forceinline__ v16h load_b_tile(const float* __restrict__ W,
                                            int k0, int Nc, int hi, int colL)
{
    // B f16 32x16 layout: element i holds K=k0+hi*8+(i<8 ? i : 16+i-8),
    // matching A; column = lane&15. Coalesced unconditional b32 loads (col clamped).
    v16h b;
    const float* wp = W + (size_t)(k0 + hi * 8) * Nc + colL;
#pragma unroll
    for (int i = 0; i < 16; ++i)
        b[i] = (_Float16)wp[(size_t)(((i >> 3) << 4) + (i & 7)) * Nc];
    return b;
}

// ---------------------------------------------------------------------------
// out[M,Nc] = A[M,K] @ W[K,Nc] + bias.  One wave -> 64 rows x 32 cols.
// M % 16 == 0 (50000 = 3125*16).
// ---------------------------------------------------------------------------
__global__ __launch_bounds__(128)
void gat_gemm_bias_wmma(const float* __restrict__ A,
                        const float* __restrict__ W,
                        const float* __restrict__ bias,
                        float* __restrict__ out,
                        int M, int K, int Nc, int tilesN)
{
    const int tilesM  = M >> 4;
    const int groupsM = (tilesM + 3) >> 2;            // 4 M-tiles per wave
    const int tilesN2 = (tilesN + 1) >> 1;            // 2 N-tiles per wave
    // Scalarize the wave id: all tile indices / guards become SALU branches,
    // keeping EXEC all-1s through every WMMA.
    const int wave = __builtin_amdgcn_readfirstlane(
        blockIdx.x * (blockDim.x >> 5) + (threadIdx.x >> 5));
    if (wave >= groupsM * tilesN2) return;

    const int gm   = wave / tilesN2;
    const int tn0  = (wave - gm * tilesN2) * 2;       // first N-tile
    const bool n1ok = (tn0 + 1) < tilesN;             // scalar
    const int lane = threadIdx.x & 31;
    const int l15  = lane & 15;
    const int hi   = lane >> 4;                       // lane group 0/1

    const int col0  = tn0 * 16 + l15;
    const int col1  = col0 + 16;
    const int colL0 = (col0 < Nc) ? col0 : (Nc - 1);  // clamped for loads
    const int colL1 = (col1 < Nc) ? col1 : (Nc - 1);
    const int tm0   = gm * 4;

    v8f acc[4][2] = {};

    const int Kmain = K & ~31;                        // 64/128/256 -> all of K
    for (int k0 = 0; k0 < Kmain; k0 += 32) {
        const v16h b0 = load_b_tile(W, k0, Nc, hi, colL0);
        v16h b1;
        if (n1ok) b1 = load_b_tile(W, k0, Nc, hi, colL1);
#pragma unroll
        for (int j = 0; j < 4; ++j) {
            const int tmj = tm0 + j;
            if (tmj < tilesM) {                       // scalar guard
                const int rowA = tmj * 16 + l15;
                const v16h a = cvt_a_fast(A + (size_t)rowA * K + k0 + hi * 8);
                acc[j][0] = __builtin_amdgcn_wmma_f32_16x16x32_f16(
                    false, a, false, b0, (short)0, acc[j][0], false, false);
                if (n1ok)
                    acc[j][1] = __builtin_amdgcn_wmma_f32_16x16x32_f16(
                        false, a, false, b1, (short)0, acc[j][1], false, false);
            }
        }
    }

    if (Kmain < K) {                                  // remainder (layer 0, K=2)
        const int k0 = Kmain;
        v16h b0, b1;
#pragma unroll
        for (int i = 0; i < 16; ++i) {
            const int k = k0 + hi * 8 + ((i >> 3) << 4) + (i & 7);
            const bool ok = (k < K);
            b0[i] = (_Float16)(ok ? W[(size_t)k * Nc + colL0] : 0.0f);
            b1[i] = (_Float16)((ok && n1ok) ? W[(size_t)k * Nc + colL1] : 0.0f);
        }
#pragma unroll
        for (int j = 0; j < 4; ++j) {
            const int tmj = tm0 + j;
            if (tmj < tilesM) {
                const int rowA = tmj * 16 + l15;
                v16h a;
#pragma unroll
                for (int i = 0; i < 16; ++i) {
                    const int k = k0 + hi * 8 + ((i >> 3) << 4) + (i & 7);
                    a[i] = (_Float16)((k < K) ? A[(size_t)rowA * K + k] : 0.0f);
                }
                acc[j][0] = __builtin_amdgcn_wmma_f32_16x16x32_f16(
                    false, a, false, b0, (short)0, acc[j][0], false, false);
                if (n1ok)
                    acc[j][1] = __builtin_amdgcn_wmma_f32_16x16x32_f16(
                        false, a, false, b1, (short)0, acc[j][1], false, false);
            }
        }
    }

    // C/D layout: VGPR q of lane -> row (tile*16 + q + hi*8), col = lane&15.
    const float bb0 = bias[colL0];
    const float bb1 = bias[colL1];
#pragma unroll
    for (int j = 0; j < 4; ++j) {
        const int tmj = tm0 + j;
        if (tmj < tilesM) {                           // scalar guard
            const int rbase = tmj * 16 + hi * 8;
            if (col0 < Nc) {
#pragma unroll
                for (int q = 0; q < 8; ++q)
                    out[(size_t)(rbase + q) * Nc + col0] = acc[j][0][q] + bb0;
            }
            if (n1ok && col1 < Nc) {
#pragma unroll
                for (int q = 0; q < 8; ++q)
                    out[(size_t)(rbase + q) * Nc + col1] = acc[j][1][q] + bb1;
            }
        }
    }
}

// ---------------------------------------------------------------------------
// Monotonic float <-> uint key for atomic segment-max (handles negatives).
// ---------------------------------------------------------------------------
__device__ __forceinline__ unsigned fkey(float f) {
    unsigned u = __float_as_uint(f);
    return (u & 0x80000000u) ? ~u : (u | 0x80000000u);
}
__device__ __forceinline__ float funkey(unsigned k) {
    return (k & 0x80000000u) ? __uint_as_float(k & 0x7fffffffu)
                             : __uint_as_float(~k);
}

// Zero the aggregation buffer, segment-max keys and softmax denominators.
__global__ void gat_init(float* __restrict__ agg, unsigned* __restrict__ mmax,
                         float* __restrict__ denom, int nAgg, int nNH)
{
    int t = blockIdx.x * blockDim.x + threadIdx.x;
    if (t < nAgg) agg[t] = 0.0f;
    if (t < nNH)  { mmax[t] = 0u; denom[t] = 0.0f; }
}

// score[e,h] = sum_c att[h,c] * leakyrelu(xl[src,h,c] + xr[dst,h,c])
// + running segment max over dst.
__global__ void gat_score(const int* __restrict__ ei,
                          const float* __restrict__ xl,
                          const float* __restrict__ xr,
                          const float* __restrict__ att,
                          float* __restrict__ score,
                          unsigned* __restrict__ mmax,
                          int H, int C)
{
    int t = blockIdx.x * blockDim.x + threadIdx.x;
    if (t >= GAT_E2 * H) return;
    const int e = t / H, h = t - e * H;
    const int src = (e < GAT_E) ? ei[e]         : (e - GAT_E);
    const int dst = (e < GAT_E) ? ei[GAT_E + e] : (e - GAT_E);
    const int HC = H * C;
    const float* pl = xl + (size_t)src * HC + h * C;
    const float* pr = xr + (size_t)dst * HC + h * C;
    const float* pa = att + h * C;
    float s = 0.0f;
    if ((C & 3) == 0) {                       // C = 16/32: b128 gathers
        for (int c = 0; c < C; c += 4) {
            const float4 vl = *(const float4*)(pl + c);
            const float4 vr = *(const float4*)(pr + c);
            const float4 va = *(const float4*)(pa + c);
            float v;
            v = vl.x + vr.x; s += ((v > 0.f) ? v : 0.2f * v) * va.x;
            v = vl.y + vr.y; s += ((v > 0.f) ? v : 0.2f * v) * va.y;
            v = vl.z + vr.z; s += ((v > 0.f) ? v : 0.2f * v) * va.z;
            v = vl.w + vr.w; s += ((v > 0.f) ? v : 0.2f * v) * va.w;
        }
    } else {
        for (int c = 0; c < C; ++c) {
            float v = pl[c] + pr[c];
            v = (v > 0.0f) ? v : 0.2f * v;    // leaky_relu, NEG_SLOPE=0.2
            s += v * pa[c];
        }
    }
    score[t] = s;
    atomicMax(&mmax[dst * H + h], fkey(s));
}

// alpha = exp(score - max[dst]);  denom[dst] += alpha  (alpha stored in-place)
__global__ void gat_alpha(const int* __restrict__ ei,
                          float* __restrict__ score,
                          const unsigned* __restrict__ mmax,
                          float* __restrict__ denom, int H)
{
    int t = blockIdx.x * blockDim.x + threadIdx.x;
    if (t >= GAT_E2 * H) return;
    const int e = t / H, h = t - e * H;
    const int dst = (e < GAT_E) ? ei[GAT_E + e] : (e - GAT_E);
    const float m = funkey(mmax[dst * H + h]);
    const float a = __expf(score[t] - m);
    score[t] = a;
    atomicAdd(&denom[dst * H + h], a);
}

// out[dst,h,:] += (alpha/denom[dst,h]) * xl[src,h,:]
__global__ void gat_aggregate(const int* __restrict__ ei,
                              const float* __restrict__ xl,
                              const float* __restrict__ alpha,
                              const float* __restrict__ denom,
                              float* __restrict__ out,
                              int H, int C)
{
    int t = blockIdx.x * blockDim.x + threadIdx.x;
    if (t >= GAT_E2 * H) return;
    const int e = t / H, h = t - e * H;
    const int src = (e < GAT_E) ? ei[e]         : (e - GAT_E);
    const int dst = (e < GAT_E) ? ei[GAT_E + e] : (e - GAT_E);
    const int HC = H * C;
    const float a = alpha[t] / denom[dst * H + h];
    const float* pl = xl + (size_t)src * HC + h * C;
    float* po = out + (size_t)dst * HC + h * C;
    if ((C & 3) == 0) {
        for (int c = 0; c < C; c += 4) {
            const float4 v = *(const float4*)(pl + c);
            atomicAdd(&po[c + 0], a * v.x);
            atomicAdd(&po[c + 1], a * v.y);
            atomicAdd(&po[c + 2], a * v.z);
            atomicAdd(&po[c + 3], a * v.w);
        }
    } else {
        for (int c = 0; c < C; ++c)
            atomicAdd(&po[c], a * pl[c]);
    }
}

// bias add + optional head-mean + optional ELU. concat: n_elems = N*H*C
// (in-place ok); mean: n_elems = N*C, reads agg, writes out.
__global__ void gat_finalize(const float* __restrict__ agg,
                             const float* __restrict__ bias,
                             float* __restrict__ out,
                             int n_elems, int H, int C, int concat, int do_elu)
{
    int t = blockIdx.x * blockDim.x + threadIdx.x;
    if (t >= n_elems) return;
    float y;
    if (concat) {
        const int HC = H * C;
        y = agg[t] + bias[t % HC];
    } else {
        const int n = t / C, c = t - n * C;
        float s = 0.0f;
        for (int h = 0; h < H; ++h) s += agg[(size_t)n * H * C + h * C + c];
        y = s / (float)H + bias[c];
    }
    if (do_elu) y = (y > 0.0f) ? y : (__expf(y) - 1.0f);
    out[t] = y;
}

// ---------------------------------------------------------------------------
// Host side: 6-layer pipeline. Inputs: d_in[0]=x (N*2 f32), d_in[1]=edge_index
// (2*E i32), then per layer (Wl, bl, Wr, br, att, bias) -> d_in[2 + 6*i + j].
// ---------------------------------------------------------------------------
extern "C" void kernel_launch(void* const* d_in, const int* in_sizes, int n_in,
                              void* d_out, int out_size, void* d_ws, size_t ws_size,
                              hipStream_t stream)
{
    (void)in_sizes; (void)n_in; (void)out_size; (void)ws_size;

    struct LayerCfg { int cin, cout, H, concat; };
    const LayerCfg layers[6] = {
        {  2, 16, 8, 1}, {128, 32, 8, 1}, {256, 32, 8, 1},
        {256, 16, 8, 1}, {128, 16, 4, 1}, { 64,  2, 1, 0}
    };

    const float* x0 = (const float*)d_in[0];
    const int*   ei = (const int*)d_in[1];

    // Workspace layout:
    //   P0: agg / next-layer x (GEMMs that read it are stream-ordered before
    //       gat_init clobbers it)   P1: xl   P2: xr   score  mmax  denom
    char* ws = (char*)d_ws;
    const size_t BIG = (size_t)GAT_N * 256 * sizeof(float);     // 51.2 MB
    float*    P0    = (float*)(ws);
    float*    P1    = (float*)(ws + BIG);
    float*    P2    = (float*)(ws + 2 * BIG);
    float*    score = (float*)(ws + 3 * BIG);
    unsigned* mmax  = (unsigned*)(ws + 3 * BIG + (size_t)GAT_E2 * 8 * 4);
    float*    denom = (float*)(ws + 3 * BIG + (size_t)GAT_E2 * 8 * 4
                                            + (size_t)GAT_N  * 8 * 4);

    for (int i = 0; i < 6; ++i) {
        const LayerCfg L = layers[i];
        const int HC = L.H * L.cout;
        const float* Wl   = (const float*)d_in[2 + 6 * i + 0];
        const float* bl   = (const float*)d_in[2 + 6 * i + 1];
        const float* Wr   = (const float*)d_in[2 + 6 * i + 2];
        const float* br   = (const float*)d_in[2 + 6 * i + 3];
        const float* att  = (const float*)d_in[2 + 6 * i + 4];
        const float* bias = (const float*)d_in[2 + 6 * i + 5];

        const float* xin = (i == 0) ? x0 : P0;

        // Dense GEMMs on the WMMA path: xl = xin@Wl + bl, xr = xin@Wr + br
        const int tilesN  = (HC + 15) / 16;
        const int tilesN2 = (tilesN + 1) / 2;                // 32 cols / wave
        const int groupsM = ((GAT_N / 16) + 3) / 4;          // 64 rows / wave
        const int waves   = groupsM * tilesN2;
        const int gblks   = (waves + 3) / 4;                 // 4 waves / block
        gat_gemm_bias_wmma<<<gblks, 128, 0, stream>>>(xin, Wl, bl, P1,
                                                      GAT_N, L.cin, HC, tilesN);
        gat_gemm_bias_wmma<<<gblks, 128, 0, stream>>>(xin, Wr, br, P2,
                                                      GAT_N, L.cin, HC, tilesN);

        // Reset agg (P0), per-(node,head) running max keys and denominators.
        const int nAgg = GAT_N * HC, nNH = GAT_N * L.H;
        gat_init<<<(nAgg + 255) / 256, 256, 0, stream>>>(P0, mmax, denom,
                                                         nAgg, nNH);

        // Edge stage: logits -> segment softmax -> weighted scatter-add.
        const int TH = GAT_E2 * L.H;
        gat_score    <<<(TH + 255) / 256, 256, 0, stream>>>(ei, P1, P2, att,
                                                            score, mmax,
                                                            L.H, L.cout);
        gat_alpha    <<<(TH + 255) / 256, 256, 0, stream>>>(ei, score, mmax,
                                                            denom, L.H);
        gat_aggregate<<<(TH + 255) / 256, 256, 0, stream>>>(ei, P1, score,
                                                            denom, P0,
                                                            L.H, L.cout);

        // Epilogue: bias (+mean for last layer), ELU on all but last layer.
        const int nFin = L.concat ? (GAT_N * HC) : (GAT_N * L.cout);
        float* dstp = (i == 5) ? (float*)d_out : P0;
        gat_finalize<<<(nFin + 255) / 256, 256, 0, stream>>>(P0, bias, dstp,
                                                             nFin, L.H, L.cout,
                                                             L.concat, i < 5);
    }
}
